// Global_Scale_Attention_82454782149059
// MI455X (gfx1250) — matrix operand order repaired and checked by reading, not compiled, not run on hardware
//
#include <hip/hip_runtime.h>
#include <hip/hip_bf16.h>

typedef __attribute__((ext_vector_type(16))) __bf16 v16bf;
typedef __attribute__((ext_vector_type(8)))  float  v8f;

#define NHD   8      // heads
#define BATCH 4
#define HH    32
#define WW    32
#define MM    21     // windows
#define CC    32     // channels
#define HD    4      // head dim per window element
#define NN    1024   // H*W tokens
#define BH    32     // BATCH*NHD
#define DP    96     // padded per-head feature dim (M*hd = 84 -> 96)
#define TDIM  63     // 2H-1
#define TSZ   (TDIM*TDIM)  // 3969

// ---------------- WMMA fragment loaders (bf16, 16x16x32) ----------------
union FragBF { v16bf v; uint4 q[2]; };

// A operand: row-major source; lane holds row (lane&15) of p,
// K columns {hi*8 .. hi*8+7} and {16+hi*8 .. 16+hi*8+7}.
__device__ __forceinline__ v16bf load_a(const __bf16* p, int stride, int lane) {
  int m = lane & 15, hi = lane >> 4;
  FragBF f;
  const __bf16* r = p + (size_t)m * stride;
  f.q[0] = *(const uint4*)(r + hi * 8);
  f.q[1] = *(const uint4*)(r + 16 + hi * 8);
  return f.v;
}
// B operand: source stored transposed (one row per output column N);
// lane holds row (lane&15), K columns {hi*8 .. hi*8+7} and {16+hi*8 .. 16+hi*8+7}.
__device__ __forceinline__ v16bf load_b(const __bf16* p, int stride, int lane) {
  int n = lane & 15, hi = lane >> 4;
  FragBF f;
  const __bf16* r = p + (size_t)n * stride;
  f.q[0] = *(const uint4*)(r + hi * 8);
  f.q[1] = *(const uint4*)(r + 16 + hi * 8);
  return f.v;
}

// ---------------- Stage A: qkv projection + scatter to WMMA layouts ------
__global__ __launch_bounds__(256)
void qkv_kernel(const float* __restrict__ x, const float* __restrict__ Wqkv,
                const float* __restrict__ bqkv,
                __bf16* __restrict__ qbuf, __bf16* __restrict__ kbuf,
                __bf16* __restrict__ vtbuf) {
  size_t tid = (size_t)blockIdx.x * blockDim.x + threadIdx.x;
  const size_t total = (size_t)BATCH * NN * MM * 96;
  if (tid >= total) return;
  int col = (int)(tid % 96);
  size_t row = tid / 96;              // bn*MM + m
  int m = (int)(row % MM);
  size_t bn = row / MM;
  int b = (int)(bn >> 10), n = (int)(bn & 1023);

  const float* xr = x + row * CC;
  const float* w  = Wqkv + col;
  float acc = bqkv[col];
#pragma unroll
  for (int k = 0; k < CC; ++k) acc += xr[k] * w[k * 96];

  int s = col >> 5, cc = col & 31;
  int h = cc >> 2, d = cc & 3;
  int bh = b * NHD + h;
  int dim = m * HD + d;
  if (s == 0)      qbuf[((size_t)bh * NN + n) * DP + dim] = (__bf16)(acc * 0.03125f); // * N^-0.5
  else if (s == 1) kbuf[((size_t)bh * NN + n) * DP + dim] = (__bf16)acc;
  else             vtbuf[((size_t)bh * DP + dim) * NN + n] = (__bf16)acc;             // transposed
}

// ---------------- LSPE: x2 squeeze -------------------------------------
__global__ __launch_bounds__(256)
void x2_kernel(const float* __restrict__ x, const float* __restrict__ Wsq,
               const float* __restrict__ bsq, float* __restrict__ x2) {
  int tid = blockIdx.x * blockDim.x + threadIdx.x;
  if (tid >= BATCH * NHD * NN) return;
  int pos = tid & (NN - 1);
  int bh = tid >> 10;
  int h = bh & 7, b = bh >> 3;
  size_t bn = (size_t)b * NN + pos;
  const float* xr = x + bn * MM * CC + h * HD;
  float acc = bsq[0];
  for (int md = 0; md < MM; ++md)
#pragma unroll
    for (int d = 0; d < HD; ++d)
      acc += xr[md * CC + d] * Wsq[md * HD + d];
  x2[tid] = acc;   // layout (b, h, y, x)
}

// ---------------- LSPE: avg+max pool per (b,h) -------------------------
__global__ __launch_bounds__(256)
void pool_kernel(const float* __restrict__ x2, float* __restrict__ p) {
  __shared__ float ss[256], sm[256];
  int bh = blockIdx.x;
  int t = threadIdx.x;
  const float* src = x2 + bh * NN;
  float s = 0.f, mx = -1e30f;
  for (int i = t; i < NN; i += 256) { float v = src[i]; s += v; mx = fmaxf(mx, v); }
  ss[t] = s; sm[t] = mx;
  __syncthreads();
  for (int off = 128; off > 0; off >>= 1) {
    if (t < off) { ss[t] += ss[t + off]; sm[t] = fmaxf(sm[t], sm[t + off]); }
    __syncthreads();
  }
  if (t == 0) p[bh] = ss[0] * (1.f / NN) + sm[0];
}

// ------- LSPE: depthwise conv + pooled, batch-mean -> x_weight, x_proj --
__global__ __launch_bounds__(256)
void wproj_kernel(const float* __restrict__ x2, const float* __restrict__ Wdw,
                  const float* __restrict__ bdw, const float* __restrict__ p,
                  float* __restrict__ xw, float* __restrict__ xp) {
  int tid = blockIdx.x * blockDim.x + threadIdx.x;
  if (tid >= NHD * NN) return;
  int pos = tid & (NN - 1);
  int h = tid >> 10;
  int y = pos >> 5, xx = pos & 31;
  int gA = h >> 1;        // input channel for out channel h      (groups=8, 2 out/group)
  int gB = (8 + h) >> 1;  // input channel for out channel 8+h
  float accA = 0.f, accB = 0.f;
  for (int b = 0; b < BATCH; ++b) {
    const float* inA = x2 + (b * NHD + gA) * NN;
    const float* inB = x2 + (b * NHD + gB) * NN;
    float cA = bdw[h], cB = bdw[8 + h];
#pragma unroll
    for (int ky = 0; ky < 3; ++ky) {
      int yy = y + ky - 1;
      if (yy < 0 || yy >= HH) continue;
#pragma unroll
      for (int kx = 0; kx < 3; ++kx) {
        int xc = xx + kx - 1;
        if (xc < 0 || xc >= WW) continue;
        cA += inA[yy * WW + xc] * Wdw[h * 9 + ky * 3 + kx];
        cB += inB[yy * WW + xc] * Wdw[(8 + h) * 9 + ky * 3 + kx];
      }
    }
    float pool = p[b * NHD + h];
    accA += cA + pool;
    accB += cB + pool;
  }
  xw[tid] = accA * 0.25f;
  xp[tid] = accB * 0.25f;
}

// ------- LSPE: full correlation -> 63x63 bias table per head -----------
__global__ __launch_bounds__(256)
void table_kernel(const float* __restrict__ xw, const float* __restrict__ xp,
                  float* __restrict__ table2d) {
  int tid = blockIdx.x * blockDim.x + threadIdx.x;
  if (tid >= NHD * TSZ) return;
  int h = tid / TSZ;
  int pq = tid % TSZ;
  int pp = pq / TDIM, qq = pq % TDIM;
  const float* Wt = xw + h * NN;
  const float* Pt = xp + h * NN;
  int u0 = pp - 31; if (u0 < 0) u0 = 0;
  int u1 = pp; if (u1 > 31) u1 = 31;
  int v0 = qq - 31; if (v0 < 0) v0 = 0;
  int v1 = qq; if (v1 > 31) v1 = 31;
  float acc = 0.f;
  for (int u = u0; u <= u1; ++u)
    for (int v = v0; v <= v1; ++v)
      acc += Pt[(pp - u) * WW + (qq - v)] * Wt[u * WW + v];
  table2d[tid] = acc;   // (h, p, q)
}

// ---------------- Fused flash attention (WMMA bf16) --------------------
// One wave32 per block; block = (bh, 16-row query tile). Online softmax.
__global__ __launch_bounds__(32)
void attn_kernel(const __bf16* __restrict__ qbuf, const __bf16* __restrict__ kbuf,
                 const __bf16* __restrict__ vtbuf, const float* __restrict__ table2d,
                 float* __restrict__ obuf) {
  __shared__ float s_table[TSZ];
  __shared__ __align__(16) __bf16 s_ptile[16 * 32];

  const int lane = threadIdx.x;
  const int blk  = blockIdx.x;
  const int bh   = blk >> 6;
  const int q0   = (blk & 63) << 4;
  const int h    = bh & (NHD - 1);

  for (int i = lane; i < TSZ; i += 32) s_table[i] = table2d[h * TSZ + i];
  __syncthreads();

  const __bf16* Q  = qbuf  + (size_t)bh * NN * DP;
  const __bf16* K  = kbuf  + (size_t)bh * NN * DP;
  const __bf16* Vt = vtbuf + (size_t)bh * DP * NN;

  v16bf qf[3];
#pragma unroll
  for (int kc = 0; kc < 3; ++kc)
    qf[kc] = load_a(Q + (size_t)q0 * DP + kc * 32, DP, lane);

  v8f oacc[6];
#pragma unroll
  for (int j = 0; j < 6; ++j) oacc[j] = {};

  float mrow[8], lrow[8];
#pragma unroll
  for (int r = 0; r < 8; ++r) { mrow[r] = -1e30f; lrow[r] = 0.f; }

  const int hi = lane >> 4;
  const int ln = lane & 15;
  const int mbase = q0 + hi * 8;

  for (int kb = 0; kb < NN; kb += 32) {
    // scores: two 16x16 C tiles over K = 96 (3 chunks of 32)
    v8f s0 = {}, s1 = {};
#pragma unroll
    for (int kc = 0; kc < 3; ++kc) {
      v16bf b0 = load_b(K + (size_t)kb * DP + kc * 32, DP, lane);
      v16bf b1 = load_b(K + (size_t)(kb + 16) * DP + kc * 32, DP, lane);
      s0 = __builtin_amdgcn_wmma_f32_16x16x32_bf16(false, qf[kc], false, b0, (short)0, s0, false, false);
      s1 = __builtin_amdgcn_wmma_f32_16x16x32_bf16(false, qf[kc], false, b1, (short)0, s1, false, false);
    }
    // dynamic relative-position bias + online softmax update
    int nk0 = kb + ln, nk1 = kb + 16 + ln;
    int yk0 = nk0 >> 5, xk0 = nk0 & 31;
    int yk1 = nk1 >> 5, xk1 = nk1 & 31;
#pragma unroll
    for (int r = 0; r < 8; ++r) {
      int nq = mbase + r;
      int yq = nq >> 5, xq = nq & 31;
      float a = s0[r] + s_table[(yq - yk0 + 31) * TDIM + (xq - xk0 + 31)];
      float b = s1[r] + s_table[(yq - yk1 + 31) * TDIM + (xq - xk1 + 31)];
      float cm = fmaxf(a, b);
#pragma unroll
      for (int off = 1; off < 16; off <<= 1) cm = fmaxf(cm, __shfl_xor(cm, off, 32));
      float mnew = fmaxf(mrow[r], cm);
      float scl  = __expf(mrow[r] - mnew);
      float ea = __expf(a - mnew);
      float eb = __expf(b - mnew);
      float ps = ea + eb;
#pragma unroll
      for (int off = 1; off < 16; off <<= 1) ps += __shfl_xor(ps, off, 32);
      lrow[r] = lrow[r] * scl + ps;
      mrow[r] = mnew;
#pragma unroll
      for (int j = 0; j < 6; ++j) oacc[j][r] *= scl;
      int m = hi * 8 + r;                       // C-layout -> row-major stage
      s_ptile[m * 32 + ln]      = (__bf16)ea;
      s_ptile[m * 32 + 16 + ln] = (__bf16)eb;
    }
    __syncthreads();
    // P @ V for this 32-key chunk (6 dim tiles of 16)
    v16bf pa = load_a(s_ptile, 32, lane);
#pragma unroll
    for (int j = 0; j < 6; ++j) {
      v16bf vb = load_b(Vt + (size_t)(j * 16) * NN + kb, NN, lane);
      oacc[j] = __builtin_amdgcn_wmma_f32_16x16x32_bf16(false, pa, false, vb, (short)0, oacc[j], false, false);
    }
    __syncthreads();
  }

  // normalize rows and store f32: obuf layout (bh, n, DP)
#pragma unroll
  for (int j = 0; j < 6; ++j) {
    int dim = j * 16 + ln;
#pragma unroll
    for (int r = 0; r < 8; ++r) {
      int n = mbase + r;
      obuf[((size_t)bh * NN + n) * DP + dim] = oacc[j][r] / lrow[r];
    }
  }
}

// ---------------- Stage C: head merge + output projection ---------------
__global__ __launch_bounds__(256)
void proj_kernel(const float* __restrict__ obuf, const float* __restrict__ Wproj,
                 const float* __restrict__ bproj, float* __restrict__ out) {
  size_t tid = (size_t)blockIdx.x * blockDim.x + threadIdx.x;
  const size_t total = (size_t)BATCH * NN * MM * CC;
  if (tid >= total) return;
  int c = (int)(tid & 31);
  size_t rm = tid >> 5;           // bn*MM + m
  int m = (int)(rm % MM);
  size_t bn = rm / MM;
  int b = (int)(bn >> 10), n = (int)(bn & 1023);
  float acc = bproj[c];
#pragma unroll
  for (int h2 = 0; h2 < NHD; ++h2) {
    const float* orow = obuf + ((size_t)(b * NHD + h2) * NN + n) * DP + m * HD;
    const float* wr = Wproj + (h2 * HD) * CC + c;
#pragma unroll
    for (int d2 = 0; d2 < HD; ++d2)
      acc += orow[d2] * wr[d2 * CC];
  }
  out[tid] = acc;
}

// ---------------- launch -------------------------------------------------
extern "C" void kernel_launch(void* const* d_in, const int* in_sizes, int n_in,
                              void* d_out, int out_size, void* d_ws, size_t ws_size,
                              hipStream_t stream) {
  const float* x     = (const float*)d_in[0];
  const float* Wqkv  = (const float*)d_in[1];
  const float* bqkv  = (const float*)d_in[2];
  const float* Wproj = (const float*)d_in[3];
  const float* bproj = (const float*)d_in[4];
  const float* Wsq   = (const float*)d_in[5];
  const float* bsq   = (const float*)d_in[6];
  const float* Wdw   = (const float*)d_in[7];
  const float* bdw   = (const float*)d_in[8];

  char* ws = (char*)d_ws;
  const size_t qkvBytes = (size_t)BH * NN * DP * sizeof(__bf16);   // 6,291,456 each
  __bf16* qbuf  = (__bf16*)(ws);
  __bf16* kbuf  = (__bf16*)(ws + qkvBytes);
  __bf16* vtbuf = (__bf16*)(ws + 2 * qkvBytes);
  float*  obuf  = (float*)(ws + 3 * qkvBytes);                     // 12,582,912
  char* ws2 = ws + 3 * qkvBytes + (size_t)BH * NN * DP * sizeof(float);
  float* x2    = (float*)(ws2);                                    // 131,072
  float* pbuf  = (float*)(ws2 + 131072);                           // 256 (padded)
  float* xw    = (float*)(ws2 + 131072 + 256);                     // 32,768
  float* xp    = (float*)(ws2 + 131072 + 256 + 32768);             // 32,768
  float* tab   = (float*)(ws2 + 131072 + 256 + 2 * 32768);         // 127,008

  // zero q/k/vt so the 84->96 K padding contributes nothing
  hipMemsetAsync(qbuf, 0, 3 * qkvBytes, stream);

  {
    size_t total = (size_t)BATCH * NN * MM * 96;             // 8,257,536
    qkv_kernel<<<(unsigned)((total + 255) / 256), 256, 0, stream>>>(
        x, Wqkv, bqkv, qbuf, kbuf, vtbuf);
  }
  x2_kernel<<<(BATCH * NHD * NN + 255) / 256, 256, 0, stream>>>(x, Wsq, bsq, x2);
  pool_kernel<<<BH, 256, 0, stream>>>(x2, pbuf);
  wproj_kernel<<<(NHD * NN + 255) / 256, 256, 0, stream>>>(x2, Wdw, bdw, pbuf, xw, xp);
  table_kernel<<<(NHD * TSZ + 255) / 256, 256, 0, stream>>>(xw, xp, tab);
  attn_kernel<<<BH * (NN / 16), 32, 0, stream>>>(qbuf, kbuf, vtbuf, tab, obuf);
  {
    size_t total = (size_t)BATCH * NN * MM * CC;             // 2,752,512
    proj_kernel<<<(unsigned)((total + 255) / 256), 256, 0, stream>>>(
        obuf, Wproj, bproj, (float*)d_out);
  }
}
